// Model_65120294142555
// MI455X (gfx1250) — matrix operand order repaired and checked by reading, not compiled, not run on hardware
//
#include <hip/hip_runtime.h>
#include <stdint.h>

typedef __attribute__((ext_vector_type(16))) __bf16 v16bf;
typedef __attribute__((ext_vector_type(8)))  float  v8f;
typedef __attribute__((ext_vector_type(4)))  unsigned int u32x4;
typedef unsigned short ushort_t;

#define B_   2
#define S_   2048
#define H_   2048
#define NH_  16
#define NKV_ 4
#define DH_  128
#define I_   8192
#define M_   (B_*S_)
#define EPS_ 1e-6f
#define SCALING_ 0.08838834764831845f   /* 1/sqrt(DH) */

static __device__ __forceinline__ ushort_t f32_to_bf16(float f) {
    union { float f; uint32_t u; } v; v.f = f;
    uint32_t r = v.u + 0x7FFFu + ((v.u >> 16) & 1u);
    return (ushort_t)(r >> 16);
}

union FragU { u32x4 q[2]; v16bf v; };

// A-fragment (16x32 bf16, MxK): lanes 0-15 row=l hold K {0..7,16..23};
// lanes 16-31 row=l-16 hold K {8..15,24..31}. Two ds_load_b128.
static __device__ __forceinline__ v16bf load_frag_a(const ushort_t* lds, int row_base,
                                                    int pitch, int k0, int lane) {
    int l = lane & 15; int hi = lane >> 4;
    const ushort_t* p = lds + (size_t)(row_base + l) * pitch + k0 + (hi ? 8 : 0);
    FragU f;
    f.q[0] = *(const u32x4*)(p);
    f.q[1] = *(const u32x4*)(p + 16);
    return f.v;
}

// B-fragment (32x16 bf16, KxN) from N-major (K-contiguous) LDS tile:
// lanes 0-15 col=l hold K {0..7,16..23}; lanes 16-31 hold K {8..15,24..31}. Two ds_load_b128.
static __device__ __forceinline__ v16bf load_frag_b(const ushort_t* lds, int col_base,
                                                    int pitch, int k0, int lane) {
    int l = lane & 15; int hi = lane >> 4;
    const ushort_t* p = lds + (size_t)(col_base + l) * pitch + k0 + (hi ? 8 : 0);
    FragU f;
    f.q[0] = *(const u32x4*)(p);
    f.q[1] = *(const u32x4*)(p + 16);
    return f.v;
}

static __device__ __forceinline__ v8f wmma_bf16(v16bf a, v16bf b, v8f c) {
    return __builtin_amdgcn_wmma_f32_16x16x32_bf16(false, a, false, b, (short)0, c, false, false);
}

// ---------------- weight convert + transpose: f32 W[K][N] -> bf16 WT[N][K] --------
// 2D grid: no integer division. Reads coalesced along N.
__global__ __launch_bounds__(256) void convT_kernel(const float* __restrict__ W,
                                                    ushort_t* __restrict__ WT,
                                                    int Kd, int Nd) {
    int n = blockIdx.x * 256 + threadIdx.x;
    int k = blockIdx.y;
    WT[(size_t)n * Kd + k] = f32_to_bf16(W[(size_t)k * Nd + n]);
}

// ---------------- row RMSNorm -> bf16 --------------------------------------------
__global__ __launch_bounds__(256) void rmsnorm_kernel(const float* __restrict__ X,
                                                      const float* __restrict__ w,
                                                      ushort_t* __restrict__ out, int ncol) {
    int row = blockIdx.x;
    const float* x = X + (size_t)row * ncol;
    float ss = 0.f;
    for (int c = threadIdx.x; c < ncol; c += 256) { float v = x[c]; ss += v * v; }
    for (int off = 1; off < 32; off <<= 1) ss += __shfl_xor(ss, off);
    __shared__ float wsum[8];
    if ((threadIdx.x & 31) == 0) wsum[threadIdx.x >> 5] = ss;
    __syncthreads();
    float tot = 0.f;
    #pragma unroll
    for (int i = 0; i < 8; i++) tot += wsum[i];
    float rms = rsqrtf(tot / (float)ncol + EPS_);
    ushort_t* o = out + (size_t)row * ncol;
    for (int c = threadIdx.x; c < ncol; c += 256) o[c] = f32_to_bf16(x[c] * rms * w[c]);
}

// ---------------- generic bf16 WMMA GEMM: C = A(MxK) * BT(NxK)^T [+ add] ----------
// 128x128 tile, K-step 64, 8 waves (2x4), wave computes 64x32 via 8 accumulators.
template <bool HASADD>
__global__ __launch_bounds__(256) void gemm_bf16_kernel(const ushort_t* __restrict__ A,
                                                        const ushort_t* __restrict__ BT,
                                                        const float* __restrict__ addsrc,
                                                        float* __restrict__ C,
                                                        int Ndim, int Kdim) {
    __shared__ ushort_t lA[128 * 72];
    __shared__ ushort_t lB[128 * 72];
    const int tid = threadIdx.x, lane = tid & 31, wave = tid >> 5;
    const int waveM = wave >> 2, waveN = wave & 3;      // 2 x 4 wave grid
    const int tileM = blockIdx.y * 128, tileN = blockIdx.x * 128;

    v8f z = {};
    v8f acc[4][2];
    #pragma unroll
    for (int i = 0; i < 4; i++) { acc[i][0] = z; acc[i][1] = z; }

    const int lrow = tid >> 1, lhalf = tid & 1;
    const ushort_t* ga = A  + (size_t)(tileM + lrow) * Kdim + lhalf * 32;
    const ushort_t* gb = BT + (size_t)(tileN + lrow) * Kdim + lhalf * 32;

    for (int k0 = 0; k0 < Kdim; k0 += 64) {
        u32x4 av[4], bv[4];
        #pragma unroll
        for (int i = 0; i < 4; i++) av[i] = *(const u32x4*)(ga + k0 + i * 8);
        #pragma unroll
        for (int i = 0; i < 4; i++) bv[i] = *(const u32x4*)(gb + k0 + i * 8);
        __syncthreads();
        #pragma unroll
        for (int i = 0; i < 4; i++) *(u32x4*)(lA + lrow * 72 + lhalf * 32 + i * 8) = av[i];
        #pragma unroll
        for (int i = 0; i < 4; i++) *(u32x4*)(lB + lrow * 72 + lhalf * 32 + i * 8) = bv[i];
        __syncthreads();
        #pragma unroll
        for (int ks = 0; ks < 64; ks += 32) {
            v16bf af[4], bf2[2];
            #pragma unroll
            for (int mi = 0; mi < 4; mi++) af[mi]  = load_frag_a(lA, waveM * 64 + mi * 16, 72, ks, lane);
            #pragma unroll
            for (int ni = 0; ni < 2; ni++) bf2[ni] = load_frag_b(lB, waveN * 32 + ni * 16, 72, ks, lane);
            #pragma unroll
            for (int mi = 0; mi < 4; mi++)
                #pragma unroll
                for (int ni = 0; ni < 2; ni++)
                    acc[mi][ni] = wmma_bf16(af[mi], bf2[ni], acc[mi][ni]);
        }
    }

    // incremental-address epilogue (no per-element mul/branch)
    const int hi = lane >> 4, c16 = lane & 15;
    #pragma unroll
    for (int mi = 0; mi < 4; mi++) {
        int row0 = tileM + waveM * 64 + mi * 16 + (hi ? 8 : 0);
        size_t base = (size_t)row0 * Ndim + tileN + waveN * 32 + c16;
        float* cp = C + base;
        const float* ap = addsrc + base;
        #pragma unroll
        for (int r = 0; r < 8; r++) {
            #pragma unroll
            for (int ni = 0; ni < 2; ni++) {
                float v = acc[mi][ni][r];
                if (HASADD) v += ap[ni * 16];
                cp[ni * 16] = v;
            }
            cp += Ndim;
            if (HASADD) ap += Ndim;
        }
    }
}

// ---------------- fused gate/up GEMM + SiLU -> bf16 -------------------------------
__global__ __launch_bounds__(256) void gateup_kernel(const ushort_t* __restrict__ A,
                                                     const ushort_t* __restrict__ GT,
                                                     const ushort_t* __restrict__ UT,
                                                     ushort_t* __restrict__ OUT,
                                                     int Ndim, int Kdim) {
    __shared__ ushort_t lA[128 * 72];
    __shared__ ushort_t lG[128 * 72];
    __shared__ ushort_t lU[128 * 72];
    const int tid = threadIdx.x, lane = tid & 31, wave = tid >> 5;
    const int waveM = wave >> 2, waveN = wave & 3;
    const int tileM = blockIdx.y * 128, tileN = blockIdx.x * 128;

    v8f z = {};
    v8f accg[4][2], accu[4][2];
    #pragma unroll
    for (int i = 0; i < 4; i++) { accg[i][0] = z; accg[i][1] = z; accu[i][0] = z; accu[i][1] = z; }

    const int lrow = tid >> 1, lhalf = tid & 1;
    const ushort_t* ga = A  + (size_t)(tileM + lrow) * Kdim + lhalf * 32;
    const ushort_t* gg = GT + (size_t)(tileN + lrow) * Kdim + lhalf * 32;
    const ushort_t* gu = UT + (size_t)(tileN + lrow) * Kdim + lhalf * 32;

    for (int k0 = 0; k0 < Kdim; k0 += 64) {
        u32x4 av[4], gv[4], uv[4];
        #pragma unroll
        for (int i = 0; i < 4; i++) av[i] = *(const u32x4*)(ga + k0 + i * 8);
        #pragma unroll
        for (int i = 0; i < 4; i++) gv[i] = *(const u32x4*)(gg + k0 + i * 8);
        #pragma unroll
        for (int i = 0; i < 4; i++) uv[i] = *(const u32x4*)(gu + k0 + i * 8);
        __syncthreads();
        #pragma unroll
        for (int i = 0; i < 4; i++) {
            *(u32x4*)(lA + lrow * 72 + lhalf * 32 + i * 8) = av[i];
            *(u32x4*)(lG + lrow * 72 + lhalf * 32 + i * 8) = gv[i];
            *(u32x4*)(lU + lrow * 72 + lhalf * 32 + i * 8) = uv[i];
        }
        __syncthreads();
        #pragma unroll
        for (int ks = 0; ks < 64; ks += 32) {
            v16bf af[4];
            #pragma unroll
            for (int mi = 0; mi < 4; mi++) af[mi] = load_frag_a(lA, waveM * 64 + mi * 16, 72, ks, lane);
            #pragma unroll
            for (int ni = 0; ni < 2; ni++) {
                v16bf bg = load_frag_b(lG, waveN * 32 + ni * 16, 72, ks, lane);
                v16bf bu = load_frag_b(lU, waveN * 32 + ni * 16, 72, ks, lane);
                #pragma unroll
                for (int mi = 0; mi < 4; mi++) {
                    accg[mi][ni] = wmma_bf16(af[mi], bg, accg[mi][ni]);
                    accu[mi][ni] = wmma_bf16(af[mi], bu, accu[mi][ni]);
                }
            }
        }
    }

    const int hi = lane >> 4, c16 = lane & 15;
    #pragma unroll
    for (int mi = 0; mi < 4; mi++) {
        int row0 = tileM + waveM * 64 + mi * 16 + (hi ? 8 : 0);
        ushort_t* op = OUT + (size_t)row0 * Ndim + tileN + waveN * 32 + c16;
        #pragma unroll
        for (int r = 0; r < 8; r++) {
            #pragma unroll
            for (int ni = 0; ni < 2; ni++) {
                float g = accg[mi][ni][r];
                float u = accu[mi][ni][r];
                float h = (g / (1.f + __expf(-g))) * u;     // SiLU(g) * u
                op[ni * 16] = f32_to_bf16(h);
            }
            op += Ndim;
        }
    }
}

// ---------------- Q/K RMSNorm + RoPE -> bf16 --------------------------------------
__global__ __launch_bounds__(128) void qknorm_rope_kernel(const float* __restrict__ q,
                                                          const float* __restrict__ k,
                                                          const float* __restrict__ cosb,
                                                          const float* __restrict__ sinb,
                                                          const float* __restrict__ qw,
                                                          const float* __restrict__ kw,
                                                          ushort_t* __restrict__ qb,
                                                          ushort_t* __restrict__ kb) {
    int m = blockIdx.x, hidx = blockIdx.y, d = threadIdx.x;
    const float* src; const float* w; ushort_t* dst;
    if (hidx < NH_) {
        src = q + ((size_t)m * NH_ + hidx) * DH_;  w = qw;
        dst = qb + ((size_t)m * NH_ + hidx) * DH_;
    } else {
        int kh = hidx - NH_;
        src = k + ((size_t)m * NKV_ + kh) * DH_;   w = kw;
        dst = kb + ((size_t)m * NKV_ + kh) * DH_;
    }
    float x = src[d];
    float ss = x * x;
    for (int off = 1; off < 32; off <<= 1) ss += __shfl_xor(ss, off);
    __shared__ float wsum[4];
    if ((threadIdx.x & 31) == 0) wsum[threadIdx.x >> 5] = ss;
    __syncthreads();
    float tot = wsum[0] + wsum[1] + wsum[2] + wsum[3];
    float xn = x * rsqrtf(tot / (float)DH_ + EPS_) * w[d];
    __shared__ float rowbuf[DH_];
    rowbuf[d] = xn;
    __syncthreads();
    float other = (d < 64) ? -rowbuf[d + 64] : rowbuf[d - 64];
    float c  = cosb[(size_t)m * DH_ + d];
    float sn = sinb[(size_t)m * DH_ + d];
    dst[d] = f32_to_bf16(xn * c + other * sn);
}

// ---------------- V: f32 (m,kh,d) -> bf16 transposed (b,kh,d,s) -------------------
__global__ __launch_bounds__(128) void vconv_kernel(const float* __restrict__ V,
                                                    ushort_t* __restrict__ VT) {
    int m = blockIdx.x, kh = blockIdx.y, d = threadIdx.x;
    int b = m >> 11, s = m & (S_ - 1);         // S_ = 2048
    float v = V[((size_t)m * NKV_ + kh) * DH_ + d];
    VT[((size_t)(b * NKV_ + kh) * DH_ + d) * S_ + s] = f32_to_bf16(v);
}

// ---------------- flash attention (GQA), 64-query tile, 64-key chunks -------------
__global__ __launch_bounds__(256) void attention_kernel(const ushort_t* __restrict__ qb,
                                                        const ushort_t* __restrict__ kb,
                                                        const ushort_t* __restrict__ vbT,
                                                        ushort_t* __restrict__ ab) {
    __shared__ ushort_t lQ[64 * 136];    // 64 q rows x DH(128), K-contiguous
    __shared__ ushort_t lK[64 * 136];    // 64 key rows x DH(128); aliased as f32 scores
    __shared__ ushort_t lV[128 * 72];    // DH(128) rows x 64 keys (v transposed)
    __shared__ ushort_t lP[64 * 72];     // 64 q rows x 64 probs (bf16)
    __shared__ float lM[64], lL[64], lScale[64];
    float* lS = (float*)lK;              // 64 x 65 f32 score tile (16640B <= 17408B)

    const int tid = threadIdx.x, lane = tid & 31, wave = tid >> 5;
    const int waveM = wave >> 2, waveN = wave & 3;
    const int qt = blockIdx.x, n = blockIdx.y, b = blockIdx.z;
    const int kh = n / (NH_ / NKV_);
    const int s0 = qt * 64;
    const int hi = lane >> 4, c16 = lane & 15;

    // load q tile once
    {
        int row = tid >> 2, quarter = tid & 3;
        const ushort_t* g = qb + ((size_t)(b * S_ + s0 + row) * NH_ + n) * DH_ + quarter * 32;
        #pragma unroll
        for (int i = 0; i < 4; i++)
            *(u32x4*)(lQ + row * 136 + quarter * 32 + i * 8) = *(const u32x4*)(g + i * 8);
    }
    if (tid < 64) { lM[tid] = -1e30f; lL[tid] = 0.f; }

    v8f z = {};
    v8f oacc[2][2];
    oacc[0][0] = z; oacc[0][1] = z; oacc[1][0] = z; oacc[1][1] = z;

    for (int j0 = 0; j0 < S_; j0 += 64) {
        __syncthreads();  // prior P*V reads done before overwriting lK/lV
        {
            int row = tid >> 2, quarter = tid & 3;   // k chunk: 64 rows x 128
            const ushort_t* g = kb + ((size_t)(b * S_ + j0 + row) * NKV_ + kh) * DH_ + quarter * 32;
            #pragma unroll
            for (int i = 0; i < 4; i++)
                *(u32x4*)(lK + row * 136 + quarter * 32 + i * 8) = *(const u32x4*)(g + i * 8);
            int drow = tid >> 1, half = tid & 1;     // vT chunk: 128 rows x 64
            const ushort_t* gv = vbT + ((size_t)(b * NKV_ + kh) * DH_ + drow) * S_ + j0 + half * 32;
            #pragma unroll
            for (int i = 0; i < 4; i++)
                *(u32x4*)(lV + drow * 72 + half * 32 + i * 8) = *(const u32x4*)(gv + i * 8);
        }
        __syncthreads();

        // scores: 64x64 tile, wave computes 32 rows x 16 cols
        v8f sacc[2]; sacc[0] = z; sacc[1] = z;
        #pragma unroll
        for (int d0 = 0; d0 < 128; d0 += 32) {
            v16bf a0 = load_frag_a(lQ, waveM * 32,      136, d0, lane);
            v16bf a1 = load_frag_a(lQ, waveM * 32 + 16, 136, d0, lane);
            v16bf bk = load_frag_b(lK, waveN * 16,      136, d0, lane);
            sacc[0] = wmma_bf16(a0, bk, sacc[0]);
            sacc[1] = wmma_bf16(a1, bk, sacc[1]);
        }
        __syncthreads();  // all waves done reading lK before aliasing as lS

        #pragma unroll
        for (int mi = 0; mi < 2; mi++)
            #pragma unroll
            for (int r = 0; r < 8; r++) {
                int row = waveM * 32 + mi * 16 + (hi ? 8 + r : r);
                int col = waveN * 16 + c16;
                lS[row * 65 + col] = sacc[mi][r] * SCALING_;
            }
        __syncthreads();

        // block-wide online softmax: thread per query row
        if (tid < 64) {
            int row = tid;
            float cmax = -1e30f;
            #pragma unroll 8
            for (int c = 0; c < 64; c++) cmax = fmaxf(cmax, lS[row * 65 + c]);
            float mold = lM[row];
            float mnew = fmaxf(mold, cmax);
            float scal = __expf(mold - mnew);
            float rsum = 0.f;
            #pragma unroll 8
            for (int c = 0; c < 64; c++) {
                float p = __expf(lS[row * 65 + c] - mnew);
                rsum += p;
                lP[row * 72 + c] = f32_to_bf16(p);
            }
            lL[row] = lL[row] * scal + rsum;
            lM[row] = mnew;
            lScale[row] = scal;
        }
        __syncthreads();

        // rescale running output, then accumulate P * V
        #pragma unroll
        for (int mi = 0; mi < 2; mi++)
            #pragma unroll
            for (int r = 0; r < 8; r++) {
                float sc = lScale[waveM * 32 + mi * 16 + (hi ? 8 + r : r)];
                oacc[mi][0][r] *= sc;
                oacc[mi][1][r] *= sc;
            }
        #pragma unroll
        for (int jk = 0; jk < 64; jk += 32) {
            v16bf p0 = load_frag_a(lP, waveM * 32,      72, jk, lane);
            v16bf p1 = load_frag_a(lP, waveM * 32 + 16, 72, jk, lane);
            #pragma unroll
            for (int nj = 0; nj < 2; nj++) {
                v16bf vv = load_frag_b(lV, waveN * 32 + nj * 16, 72, jk, lane);
                oacc[0][nj] = wmma_bf16(p0, vv, oacc[0][nj]);
                oacc[1][nj] = wmma_bf16(p1, vv, oacc[1][nj]);
            }
        }
    }
    __syncthreads();

    #pragma unroll
    for (int mi = 0; mi < 2; mi++)
        #pragma unroll
        for (int nj = 0; nj < 2; nj++)
            #pragma unroll
            for (int r = 0; r < 8; r++) {
                int rloc = waveM * 32 + mi * 16 + (hi ? 8 + r : r);
                int srow = s0 + rloc;
                int d = waveN * 32 + nj * 16 + c16;
                float val = oacc[mi][nj][r] / lL[rloc];
                ab[((size_t)(b * S_ + srow) * NH_ + n) * DH_ + d] = f32_to_bf16(val);
            }
}

// ----------------------------------------------------------------------------------
extern "C" void kernel_launch(void* const* d_in, const int* in_sizes, int n_in,
                              void* d_out, int out_size, void* d_ws, size_t ws_size,
                              hipStream_t stream) {
    (void)in_sizes; (void)n_in; (void)out_size; (void)ws_size;
    const float* hs   = (const float*)d_in[0];
    const float* cosb = (const float*)d_in[1];
    const float* sinb = (const float*)d_in[2];
    const float* wq   = (const float*)d_in[3];
    const float* wk   = (const float*)d_in[4];
    const float* wv   = (const float*)d_in[5];
    const float* wo   = (const float*)d_in[6];
    const float* qnw  = (const float*)d_in[7];
    const float* knw  = (const float*)d_in[8];
    const float* anw  = (const float*)d_in[9];
    const float* mnw  = (const float*)d_in[10];
    const float* wg   = (const float*)d_in[11];
    const float* wu   = (const float*)d_in[12];
    const float* wd   = (const float*)d_in[13];
    float* out = (float*)d_out;

    char* wsp = (char*)d_ws;
    size_t off = 0;
    auto alloc = [&](size_t bytes) -> char* {
        char* p = wsp + off;
        off += (bytes + 255) & ~(size_t)255;
        return p;
    };
    ushort_t* xb  = (ushort_t*)alloc((size_t)M_ * H_ * 2);
    ushort_t* wqT = (ushort_t*)alloc((size_t)H_ * H_ * 2);
    ushort_t* wkT = (ushort_t*)alloc((size_t)H_ * NKV_ * DH_ * 2);
    ushort_t* wvT = (ushort_t*)alloc((size_t)H_ * NKV_ * DH_ * 2);
    ushort_t* woT = (ushort_t*)alloc((size_t)H_ * H_ * 2);
    ushort_t* wgT = (ushort_t*)alloc((size_t)H_ * I_ * 2);
    ushort_t* wuT = (ushort_t*)alloc((size_t)H_ * I_ * 2);
    ushort_t* wdT = (ushort_t*)alloc((size_t)I_ * H_ * 2);
    float*    qf  = (float*)alloc((size_t)M_ * NH_ * DH_ * 4);
    float*    kf  = (float*)alloc((size_t)M_ * NKV_ * DH_ * 4);
    float*    vf  = (float*)alloc((size_t)M_ * NKV_ * DH_ * 4);
    ushort_t* qbb = (ushort_t*)alloc((size_t)M_ * NH_ * DH_ * 2);
    ushort_t* kbb = (ushort_t*)alloc((size_t)M_ * NKV_ * DH_ * 2);
    ushort_t* vbT = (ushort_t*)alloc((size_t)M_ * NKV_ * DH_ * 2);
    ushort_t* ab  = (ushort_t*)alloc((size_t)M_ * H_ * 2);
    float*    h1  = (float*)alloc((size_t)M_ * H_ * 4);
    ushort_t* gu  = (ushort_t*)alloc((size_t)M_ * I_ * 2);

    // bf16-transpose all weights (after this, all GEMM weight traffic is L2-resident:
    // 116 MB bf16 total vs 192 MB L2)
    convT_kernel<<<dim3((NH_ * DH_) / 256, H_), 256, 0, stream>>>(wq, wqT, H_, NH_ * DH_);
    convT_kernel<<<dim3((NKV_ * DH_) / 256, H_), 256, 0, stream>>>(wk, wkT, H_, NKV_ * DH_);
    convT_kernel<<<dim3((NKV_ * DH_) / 256, H_), 256, 0, stream>>>(wv, wvT, H_, NKV_ * DH_);
    convT_kernel<<<dim3(H_ / 256, H_), 256, 0, stream>>>(wo, woT, H_, H_);
    convT_kernel<<<dim3(I_ / 256, H_), 256, 0, stream>>>(wg, wgT, H_, I_);
    convT_kernel<<<dim3(I_ / 256, H_), 256, 0, stream>>>(wu, wuT, H_, I_);
    convT_kernel<<<dim3(H_ / 256, I_), 256, 0, stream>>>(wd, wdT, I_, H_);

    // attention half
    rmsnorm_kernel<<<M_, 256, 0, stream>>>(hs, anw, xb, H_);
    gemm_bf16_kernel<false><<<dim3(NH_ * DH_ / 128, M_ / 128), 256, 0, stream>>>(xb, wqT, nullptr, qf, NH_ * DH_, H_);
    gemm_bf16_kernel<false><<<dim3(NKV_ * DH_ / 128, M_ / 128), 256, 0, stream>>>(xb, wkT, nullptr, kf, NKV_ * DH_, H_);
    gemm_bf16_kernel<false><<<dim3(NKV_ * DH_ / 128, M_ / 128), 256, 0, stream>>>(xb, wvT, nullptr, vf, NKV_ * DH_, H_);
    qknorm_rope_kernel<<<dim3(M_, NH_ + NKV_), 128, 0, stream>>>(qf, kf, cosb, sinb, qnw, knw, qbb, kbb);
    vconv_kernel<<<dim3(M_, NKV_), 128, 0, stream>>>(vf, vbT);
    attention_kernel<<<dim3(S_ / 64, NH_, B_), 256, 0, stream>>>(qbb, kbb, vbT, ab);
    gemm_bf16_kernel<true><<<dim3(H_ / 128, M_ / 128), 256, 0, stream>>>(ab, woT, hs, h1, H_, H_);

    // MLP half
    rmsnorm_kernel<<<M_, 256, 0, stream>>>(h1, mnw, xb, H_);
    gateup_kernel<<<dim3(I_ / 128, M_ / 128), 256, 0, stream>>>(xb, wgT, wuT, gu, I_, H_);
    gemm_bf16_kernel<true><<<dim3(H_ / 128, M_ / 128), 256, 0, stream>>>(gu, wdT, h1, out, H_, I_);
}
